// CentroidCrop_12163347383212
// MI455X (gfx1250) — matrix operand order repaired and checked by reading, not compiled, not run on hardware
//
#include <hip/hip_runtime.h>

typedef _Float16 v16h __attribute__((ext_vector_type(16)));
typedef _Float16 v8h  __attribute__((ext_vector_type(8)));
typedef float    v8f  __attribute__((ext_vector_type(8)));

#define BATCH  16
#define HF     1024
#define WF     1024
#define HS     512
#define WSZ    512
#define HC     256
#define WC     256
#define HP     260              // padded conv1-activation height (halo 2)
#define WP     260              // padded conv1-activation width  (halo 2)
#define KP     32
#define CROPSZ 160
#define PTHRESH 0.2f

// ---- workspace layout (bytes, all 256-aligned) ----
#define OFF_IMGS   0ull
#define BYTES_IMGS (16ull*512*512*4)            // 16,777,216  resized imgs f32
#define OFF_H      (OFF_IMGS + BYTES_IMGS)
#define BYTES_H    (16ull*HP*WP*32*2)           // 69,222,400  conv1 act f16, padded NHWC
#define OFF_CMS    (OFF_H + BYTES_H)
#define BYTES_CMS  (16ull*256*256*4)            // 4,194,304   confmaps f32
#define OFF_SCORES (OFF_CMS + BYTES_CMS)
#define OFF_VALS   (OFF_SCORES + BYTES_CMS)
#define OFF_IDXS   (OFF_VALS + 2048)
#define OFF_CX     (OFF_IDXS + 2048)
#define OFF_CY     (OFF_CX + 2048)
#define OFF_VF     (OFF_CY + 2048)

// ---- out layout (floats) ----
#define OUT_CROPS  0
#define OUT_OFFS   (16*32*160*160)              // 13,107,200
#define OUT_VALSO  (OUT_OFFS + 16*32*2)
#define OUT_VALID  (OUT_VALSO + 16*32)

__device__ __forceinline__ int iclamp(int v, int lo, int hi) {
  return v < lo ? lo : (v > hi ? hi : v);
}

// K0: jax.image.resize bilinear antialias 0.5x == separable 4-tap [1,3,3,1]/8
__global__ void k_resize(const float* __restrict__ in, float* __restrict__ imgs) {
  int t = blockIdx.x * 256 + threadIdx.x;            // < 16*512*512
  int x = t % WSZ, y = (t / WSZ) % HS, b = t / (WSZ * HS);
  const float wgt[4] = {0.125f, 0.375f, 0.375f, 0.125f};
  const float* src = in + (size_t)b * HF * WF;
  float acc = 0.f;
  #pragma unroll
  for (int a = 0; a < 4; ++a) {
    int yy = iclamp(2 * y - 1 + a, 0, HF - 1);
    float ra = 0.f;
    #pragma unroll
    for (int c = 0; c < 4; ++c) {
      int xx = iclamp(2 * x - 1 + c, 0, WF - 1);
      ra += wgt[c] * src[(size_t)yy * WF + xx];
    }
    acc += wgt[a] * ra;
  }
  imgs[t] = acc;
}

// K0b: zero the 2-wide halo of the padded activation tensor
__global__ void k_hpad(_Float16* __restrict__ hbuf) {
  int t = blockIdx.x * 256 + threadIdx.x;            // over 16*260*260
  if (t >= BATCH * HP * WP) return;
  int x = t % WP, y = (t / WP) % HP;
  if (x >= 2 && x < WP - 2 && y >= 2 && y < HP - 2) return;
  _Float16* p = hbuf + (size_t)t * 32;
  v8h z = {};
  *reinterpret_cast<v8h*>(p)      = z;
  *reinterpret_cast<v8h*>(p + 8)  = z;
  *reinterpret_cast<v8h*>(p + 16) = z;
  *reinterpret_cast<v8h*>(p + 24) = z;
}

// K1: conv 5x5 /2 SAME, 1->32ch, +bias, ReLU, store f16 into padded NHWC
__global__ void k_conv1(const float* __restrict__ imgs, const float* __restrict__ w1,
                        const float* __restrict__ b1, _Float16* __restrict__ hbuf) {
  __shared__ float sw1[800];
  __shared__ float sb1[32];
  int tid = threadIdx.x;
  for (int i = tid; i < 800; i += 256) sw1[i] = w1[i];
  if (tid < 32) sb1[tid] = b1[tid];
  __syncthreads();
  int t = blockIdx.x * 256 + tid;                    // < 16*256*256
  int x = t % WC, y = (t / WC) % HC, b = t / (WC * HC);
  float r[32];
  #pragma unroll
  for (int oc = 0; oc < 32; ++oc) r[oc] = sb1[oc];
  const float* src = imgs + (size_t)b * HS * WSZ;
  for (int ky = 0; ky < 5; ++ky) {
    int iy = 2 * y + ky - 1;
    if (iy < 0 || iy >= HS) continue;
    for (int kx = 0; kx < 5; ++kx) {
      int ix = 2 * x + kx - 1;
      if (ix < 0 || ix >= WSZ) continue;
      float v = src[(size_t)iy * WSZ + ix];
      const float* wr = &sw1[(ky * 5 + kx) * 32];
      #pragma unroll
      for (int oc = 0; oc < 32; ++oc) r[oc] = fmaf(v, wr[oc], r[oc]);
    }
  }
  _Float16* hp = hbuf + (size_t)((b * HP + y + 2) * WP + x + 2) * 32;
  #pragma unroll
  for (int q = 0; q < 4; ++q) {
    v8h pk;
    #pragma unroll
    for (int j = 0; j < 8; ++j) {
      float u = r[q * 8 + j];
      pk[j] = (_Float16)(u > 0.f ? u : 0.f);
    }
    *reinterpret_cast<v8h*>(hp + q * 8) = pk;
  }
}

// K2: conv 5x5 /1 SAME, 32->1ch, +bias, sigmoid via v_wmma_f32_16x16x32_f16.
// One wave -> 16 consecutive x outputs of one row. 25 taps fully unrolled:
// A loads are immediate-offset b128s from one base (spatial zero-halo in hbuf),
// B column-0 weights come from LDS via a branch-free per-lane stride (zero page
// for lanes != 0,16), so EXEC stays all-ones with no per-tap exec masking.
__global__ void k_conv2_wmma(const _Float16* __restrict__ hbuf, const float* __restrict__ w2,
                             const float* __restrict__ b2, float* __restrict__ cms) {
  __shared__ __align__(32) _Float16 sw2z[16 + 800];  // [0..15] zeros, then w2 (f16)
  int tid = threadIdx.x;
  if (tid < 16) sw2z[tid] = (_Float16)0.f;
  for (int i = tid; i < 800; i += 256) sw2z[16 + i] = (_Float16)w2[i];
  __syncthreads();
  int wave = (blockIdx.x * 256 + tid) >> 5;          // < 65536
  int lane = tid & 31;
  int x0 = (wave & 15) << 4;
  int y  = (wave >> 4) & 255;
  int b  = wave >> 12;
  int m  = lane & 15;                                // A-row (output pixel in tile)
  int cb = (lane < 16) ? 0 : 8;                      // A K-chunks {cb..cb+7, cb+16..cb+23}
  bool act = (m == 0);                               // lanes 0 / 16 carry B column 0
  int bbase = act ? (16 + ((lane >> 4) << 3)) : 0;   // 16 (K0..7,16..23) / 24 (K8..15,24..31) / zero page
  int bstep = act ? 32 : 0;
  const _Float16* abase =
      hbuf + (size_t)((b * HP + y) * WP + x0 + m) * 32 + cb;
  float b2v = b2[0];
  v8f acc = {};
  #pragma unroll
  for (int dy = 0; dy < 5; ++dy) {
    #pragma unroll
    for (int dx = 0; dx < 5; ++dx) {
      const int tap = dy * 5 + dx;
      const _Float16* hp = abase + (size_t)(dy * WP + dx) * 32;
      v8h alo = *reinterpret_cast<const v8h*>(hp);
      v8h ahi = *reinterpret_cast<const v8h*>(hp + 16);
      const _Float16* bp = &sw2z[bbase + tap * bstep];
      v8h blo = *reinterpret_cast<const v8h*>(bp);
      v8h bhi = *reinterpret_cast<const v8h*>(bp + 16);
      v16h a, bm;
      #pragma unroll
      for (int i = 0; i < 8; ++i) {
        a[i] = alo[i];  a[8 + i] = ahi[i];
        bm[i] = blo[i]; bm[8 + i] = bhi[i];
      }
      acc = __builtin_amdgcn_wmma_f32_16x16x32_f16(
          false, a, false, bm, (short)0, acc, false, false);
    }
  }
  // D column 0: lane 0 holds M=0..7 in acc[0..7], lane 16 holds M=8..15
  if (m == 0) {
    int mb = (lane == 0) ? 0 : 8;
    float* co = cms + (size_t)((b * HC + y) * WC) + x0 + mb;
    #pragma unroll
    for (int r = 0; r < 8; ++r) {
      float v = acc[r] + b2v;
      co[r] = 1.0f / (1.0f + __expf(-v));
    }
  }
}

// K3: 3x3 NMS (SAME, -inf outside) + threshold -> scores (-inf for non-peaks)
__global__ void k_nms(const float* __restrict__ cms, float* __restrict__ scores) {
  int t = blockIdx.x * 256 + threadIdx.x;            // < 16*256*256
  int x = t % WC, y = (t / WC) % HC, b = t / (WC * HC);
  const float* c0 = cms + (size_t)b * HC * WC;
  float c = c0[y * WC + x];
  float mx = -INFINITY;
  #pragma unroll
  for (int dy = -1; dy <= 1; ++dy) {
    int yy = y + dy;
    if (yy < 0 || yy >= HC) continue;
    #pragma unroll
    for (int dx = -1; dx <= 1; ++dx) {
      int xx = x + dx;
      if (xx < 0 || xx >= WC) continue;
      float v = c0[yy * WC + xx];
      mx = v > mx ? v : mx;
    }
  }
  bool peak = (c >= mx) && (c > PTHRESH);
  scores[t] = peak ? c : -INFINITY;
}

// K4: deterministic top-32 per batch. 32 passes; pass k selects the max element
// strictly after the previous pick in (value desc, index asc) order. No mutation.
__global__ void k_topk(const float* __restrict__ scores, float* __restrict__ vals,
                       int* __restrict__ idxs) {
  __shared__ float sv[1024];
  __shared__ int   si[1024];
  __shared__ float s_pv;
  __shared__ int   s_pi;
  int b = blockIdx.x, tid = threadIdx.x;
  const float* s = scores + (size_t)b * HC * WC;
  float pv = INFINITY; int pi = -1;
  for (int k = 0; k < KP; ++k) {
    float bv = -INFINITY; int bi = HC * WC;
    for (int e = tid; e < HC * WC; e += 1024) {
      float v = s[e];
      bool rem = (v < pv) || (v == pv && e > pi);
      if (rem && (v > bv || (v == bv && e < bi))) { bv = v; bi = e; }
    }
    sv[tid] = bv; si[tid] = bi;
    __syncthreads();
    for (int st = 512; st > 0; st >>= 1) {
      if (tid < st) {
        float ov = sv[tid + st]; int oi = si[tid + st];
        if (ov > sv[tid] || (ov == sv[tid] && oi < si[tid])) { sv[tid] = ov; si[tid] = oi; }
      }
      __syncthreads();
    }
    if (tid == 0) {
      vals[b * KP + k] = sv[0];
      idxs[b * KP + k] = (si[0] >= HC * WC) ? 0 : si[0];
      s_pv = sv[0]; s_pi = si[0];
    }
    __syncthreads();
    pv = s_pv; pi = s_pi;
    __syncthreads();
  }
}

// K5: 5x5 integral refinement + small outputs (offsets, vals, valid) + cx/cy for crops
__global__ void k_refine(const float* __restrict__ cms, const float* __restrict__ vals,
                         const int* __restrict__ idxs, float* __restrict__ out,
                         float* __restrict__ cxw, float* __restrict__ cyw,
                         float* __restrict__ vfw) {
  int t = blockIdx.x * 256 + threadIdx.x;
  if (t >= BATCH * KP) return;
  int b = t / KP;
  float val = vals[t];
  int idx = idxs[t];
  bool valid = val > -3.0e38f;
  int py = idx / WC, px = idx % WC;
  const float* c0 = cms + (size_t)b * HC * WC;
  float gv = 1e-12f, sdx = 0.f, sdy = 0.f;
  #pragma unroll
  for (int a = 0; a < 5; ++a) {
    int yy = py + a - 2;
    bool yin = (yy >= 0 && yy < HC);
    int yc = iclamp(yy, 0, HC - 1);
    #pragma unroll
    for (int c2 = 0; c2 < 5; ++c2) {
      int xx = px + c2 - 2;
      bool xin = (xx >= 0 && xx < WC);
      int xc = iclamp(xx, 0, WC - 1);
      float p = (yin && xin) ? c0[yc * WC + xc] : 0.f;
      gv  += p;
      sdx += p * (float)(c2 - 2);
      sdy += p * (float)(a - 2);
    }
  }
  float cx = ((float)px + sdx / gv) * 4.0f;          // * CM_STRIDE / SCALE
  float cy = ((float)py + sdy / gv) * 4.0f;
  if (!valid) { cx = 80.f; cy = 80.f; }
  out[OUT_OFFS + t * 2 + 0] = cx - 80.f;
  out[OUT_OFFS + t * 2 + 1] = cy - 80.f;
  out[OUT_VALSO + t] = valid ? val : 0.f;
  out[OUT_VALID + t] = valid ? 1.f : 0.f;
  cxw[t] = cx; cyw[t] = cy; vfw[t] = valid ? 1.f : 0.f;
}

// K6: 160x160 bilinear crops from full-res images, zero outside / invalid
__global__ void k_crop(const float* __restrict__ full, const float* __restrict__ cxw,
                       const float* __restrict__ cyw, const float* __restrict__ vfw,
                       float* __restrict__ out) {
  size_t t = (size_t)blockIdx.x * 256 + threadIdx.x; // < 13,107,200
  int j  = (int)(t % CROPSZ);
  int i  = (int)((t / CROPSZ) % CROPSZ);
  int pk = (int)(t / ((size_t)CROPSZ * CROPSZ));     // b*32 + k
  int b  = pk / KP;
  float cx = cxw[pk], cy = cyw[pk], vf = vfw[pk];
  float sy = cy - 79.5f + (float)i;
  float sx = cx - 79.5f + (float)j;
  float fy = floorf(sy), fx = floorf(sx);
  float wy = sy - fy, wx = sx - fx;
  int y0i = (int)fy, x0i = (int)fx;
  int yi0 = iclamp(y0i, 0, HF - 1), yi1 = iclamp(y0i + 1, 0, HF - 1);
  int xi0 = iclamp(x0i, 0, WF - 1), xi1 = iclamp(x0i + 1, 0, WF - 1);
  const float* src = full + (size_t)b * HF * WF;
  float v00 = src[(size_t)yi0 * WF + xi0], v01 = src[(size_t)yi0 * WF + xi1];
  float v10 = src[(size_t)yi1 * WF + xi0], v11 = src[(size_t)yi1 * WF + xi1];
  float top = v00 * (1.f - wx) + v01 * wx;
  float bot = v10 * (1.f - wx) + v11 * wx;
  float v = top * (1.f - wy) + bot * wy;
  bool inr = (sy >= 0.f && sy <= (float)(HF - 1) && sx >= 0.f && sx <= (float)(WF - 1));
  out[OUT_CROPS + t] = v * (inr ? 1.f : 0.f) * vf;
}

extern "C" void kernel_launch(void* const* d_in, const int* in_sizes, int n_in,
                              void* d_out, int out_size, void* d_ws, size_t ws_size,
                              hipStream_t stream) {
  (void)in_sizes; (void)n_in; (void)out_size; (void)ws_size;
  const float* full = (const float*)d_in[0];
  const float* w1   = (const float*)d_in[1];
  const float* b1   = (const float*)d_in[2];
  const float* w2   = (const float*)d_in[3];
  const float* b2   = (const float*)d_in[4];
  float* out = (float*)d_out;
  char* ws = (char*)d_ws;
  float*    imgs   = (float*)(ws + OFF_IMGS);
  _Float16* hbuf   = (_Float16*)(ws + OFF_H);
  float*    cms    = (float*)(ws + OFF_CMS);
  float*    scores = (float*)(ws + OFF_SCORES);
  float*    vals   = (float*)(ws + OFF_VALS);
  int*      idxs   = (int*)(ws + OFF_IDXS);
  float*    cxw    = (float*)(ws + OFF_CX);
  float*    cyw    = (float*)(ws + OFF_CY);
  float*    vfw    = (float*)(ws + OFF_VF);

  hipLaunchKernelGGL(k_resize,     dim3(16384), dim3(256),  0, stream, full, imgs);
  hipLaunchKernelGGL(k_hpad,       dim3(4225),  dim3(256),  0, stream, hbuf);
  hipLaunchKernelGGL(k_conv1,      dim3(4096),  dim3(256),  0, stream, imgs, w1, b1, hbuf);
  hipLaunchKernelGGL(k_conv2_wmma, dim3(8192),  dim3(256),  0, stream, hbuf, w2, b2, cms);
  hipLaunchKernelGGL(k_nms,        dim3(4096),  dim3(256),  0, stream, cms, scores);
  hipLaunchKernelGGL(k_topk,       dim3(16),    dim3(1024), 0, stream, scores, vals, idxs);
  hipLaunchKernelGGL(k_refine,     dim3(2),     dim3(256),  0, stream, cms, vals, idxs, out, cxw, cyw, vfw);
  hipLaunchKernelGGL(k_crop,       dim3(51200), dim3(256),  0, stream, full, cxw, cyw, vfw, out);
}
